// MultiHeadAttention_61478161875040
// MI455X (gfx1250) — matrix operand order repaired and checked by reading, not compiled, not run on hardware
//
#include <hip/hip_runtime.h>
#include <hip/hip_bf16.h>

typedef __attribute__((ext_vector_type(16))) __bf16 bf16x16;
typedef __attribute__((ext_vector_type(8)))  __bf16 bf16x8;
typedef __attribute__((ext_vector_type(8)))  float  f32x8;

#define DEV static __device__ __forceinline__

constexpr int BB = 4, S = 2048, D = 512, H = 8, A = 64;
constexpr int BH = BB * H;
constexpr long long BHSA = (long long)BB * H * S * A;

DEV f32x8 wmma_bf16(bf16x16 a, bf16x16 b, f32x8 c) {
  return __builtin_amdgcn_wmma_f32_16x16x32_bf16(false, a, false, b, (short)0, c,
                                                 false, false);
}

DEV unsigned pack_bf16(float lo, float hi) {
  union { __bf16 h[2]; unsigned u; } u_;
  u_.h[0] = (__bf16)lo; u_.h[1] = (__bf16)hi;
  return u_.u;
}

// ---- CDNA5 async global->LDS copy (ASYNCcnt path), 16B per lane ----
DEV void cp_async16(void* lds_dst, const void* gsrc) {
#if __has_builtin(__builtin_amdgcn_global_load_async_to_lds_b128)
  typedef int v4i __attribute__((vector_size(16)));
  __builtin_amdgcn_global_load_async_to_lds_b128(
      (__attribute__((address_space(1))) v4i*)gsrc,
      (__attribute__((address_space(3))) v4i*)lds_dst, 0, 0);
#else
  *(bf16x8*)lds_dst = *(const bf16x8*)gsrc;
#endif
}

DEV void wait_async() {
#if __has_builtin(__builtin_amdgcn_global_load_async_to_lds_b128)
#if __has_builtin(__builtin_amdgcn_s_wait_asynccnt)
  __builtin_amdgcn_s_wait_asynccnt(0);
#else
  asm volatile("s_wait_asynccnt 0x0" ::: "memory");
#endif
#endif
}

// A fragment 16x32 bf16 from LDS, row-major [M][K], stride ld (ld%8==0).
DEV bf16x16 load_a(const __bf16* p0, int ld) {
  int lane = threadIdx.x & 31;
  int m = lane & 15, g = lane >> 4;
  const __bf16* p = p0 + m * ld + g * 8;
  bf16x8 lo = *(const bf16x8*)(p);
  bf16x8 hi = *(const bf16x8*)(p + 16);
  bf16x16 a;
#pragma unroll
  for (int i = 0; i < 8; ++i) { a[i] = lo[i]; a[8 + i] = hi[i]; }
  return a;
}

// B fragment 32x16 bf16 from LDS stored N-major ([N][K], stride ld, ld%8==0).
DEV bf16x16 load_bt(const __bf16* p0, int ld) {
  int lane = threadIdx.x & 31;
  int n = lane & 15, g = lane >> 4;
  const __bf16* p = p0 + n * ld + g * 8;
  bf16x8 lo = *(const bf16x8*)(p);
  bf16x8 hi = *(const bf16x8*)(p + 16);
  bf16x16 b;
#pragma unroll
  for (int i = 0; i < 8; ++i) { b[i] = lo[i]; b[8 + i] = hi[i]; }
  return b;
}

// ---------------- Kernel 1: per-head projection  out = (X @ W + bias) * scale
DEV void k1_load(const float* xb, const float* wh, int rowbase, int kb, int tid,
                 float4* xr, float (*wr)[4]) {
#pragma unroll
  for (int j = 0; j < 4; ++j) {
    int c = tid + 256 * j;              // 1024 chunks of 4 floats
    int r = c >> 3, k = (c & 7) * 4;
    xr[j] = *(const float4*)(xb + (long long)(rowbase + r) * D + kb + k);
  }
#pragma unroll
  for (int j = 0; j < 2; ++j) {
    int gi = tid + 256 * j;             // 512 groups, 4 strided along k
    int n = gi >> 3, k = (gi & 7) * 4;
    const float* wp = wh + (long long)(kb + k) * A + n;
    wr[j][0] = wp[0]; wr[j][1] = wp[A]; wr[j][2] = wp[2 * A]; wr[j][3] = wp[3 * A];
  }
}

__global__ __launch_bounds__(256)
void qkv_proj_kernel(const float* __restrict__ x, const float* __restrict__ w,
                     const float* __restrict__ bias, __bf16* __restrict__ out,
                     float scale) {
  __shared__ __align__(16) __bf16 xs[2][128 * 32];
  __shared__ __align__(16) __bf16 wt[2][64 * 32];
  int tid = threadIdx.x;
  int lane = tid & 31, wave = tid >> 5;
  int nl = lane & 15, g = lane >> 4;
  int rowbase = blockIdx.x * 128;
  int bh = blockIdx.y;
  int b = bh / H, h = bh % H;
  const float* xb = x + (long long)b * S * D;
  const float* wh = w + (long long)h * D * A;
  const float* bh_ = bias + h * A;

  f32x8 acc[4];
#pragma unroll
  for (int t = 0; t < 4; ++t) acc[t] = {};

  float4 xr[4];
  float wr[2][4];
  k1_load(xb, wh, rowbase, 0, tid, xr, wr);

  constexpr int NC = D / 32;
  for (int c = 0; c < NC; ++c) {
    __bf16* xsc = xs[c & 1];
    __bf16* wtc = wt[c & 1];
    // store prefetched regs -> LDS (packed bf16 b64 stores)
#pragma unroll
    for (int j = 0; j < 4; ++j) {
      int cc = tid + 256 * j;
      int r = cc >> 3, k = (cc & 7) * 4;
      unsigned p0 = pack_bf16(xr[j].x, xr[j].y);
      unsigned p1 = pack_bf16(xr[j].z, xr[j].w);
      *(unsigned long long*)(xsc + r * 32 + k) =
          ((unsigned long long)p1 << 32) | p0;
    }
#pragma unroll
    for (int j = 0; j < 2; ++j) {
      int gi = tid + 256 * j;
      int n = gi >> 3, k = (gi & 7) * 4;
      unsigned p0 = pack_bf16(wr[j][0], wr[j][1]);
      unsigned p1 = pack_bf16(wr[j][2], wr[j][3]);
      *(unsigned long long*)(wtc + n * 32 + k) =
          ((unsigned long long)p1 << 32) | p0;
    }
    __syncthreads();
    if (c + 1 < NC) k1_load(xb, wh, rowbase, (c + 1) * 32, tid, xr, wr);
    bf16x16 af = load_a(xsc + wave * 16 * 32, 32);
#pragma unroll
    for (int t = 0; t < 4; ++t)
      acc[t] = wmma_bf16(af, load_bt(wtc + t * 16 * 32, 32), acc[t]);
  }

  __bf16* op = out + (long long)bh * S * A;
#pragma unroll
  for (int t = 0; t < 4; ++t) {
    int col = t * 16 + nl;
    float bval = bh_[col];
#pragma unroll
    for (int r = 0; r < 8; ++r) {
      int row = rowbase + wave * 16 + g * 8 + r;
      op[(long long)row * A + col] = (__bf16)((acc[t][r] + bval) * scale);
    }
  }
}

// ---------------- Kernel 2: flash attention, faithful lower-tri(-incl-diag) -1e9 mask
__global__ __launch_bounds__(256)
void attn_kernel(const __bf16* __restrict__ Q, const __bf16* __restrict__ K,
                 const __bf16* __restrict__ V, __bf16* __restrict__ O) {
  __shared__ __align__(16) __bf16 qs[128 * 64];
  __shared__ __align__(16) __bf16 kt[2][64 * 64];  // [key][a] == B^T for QK^T
  __shared__ __align__(16) __bf16 vt[2][64 * 64];  // [a][key] == B^T for PV
  __shared__ __align__(16) __bf16 ps[8][16 * 64];  // per-wave P staging
  int tid = threadIdx.x, lane = tid & 31, wave = tid >> 5;
  int nl = lane & 15, g = lane >> 4;
  int q0 = blockIdx.x * 128;
  long long base = (long long)blockIdx.y * S * A;
  const __bf16* Qp = Q + base;
  const __bf16* Kp = K + base;
  const __bf16* Vp = V + base;

  // prologue: async-stage Q tile (contiguous 16KB) + K tile 0 (8KB), V tile 0 regs
  const __bf16* Qtile = Qp + (long long)q0 * A;
#pragma unroll
  for (int j = 0; j < 4; ++j) {
    int c = tid + 256 * j;
    cp_async16(qs + c * 8, Qtile + c * 8);
  }
#pragma unroll
  for (int j = 0; j < 2; ++j) {
    int c = tid + 256 * j;
    cp_async16(kt[0] + c * 8, Kp + c * 8);
  }
  int vr = (tid >> 3) * 2;                 // even rows 0..62 of V tile
  int vc8 = (tid & 7) * 8;
  bf16x8 ra = *(const bf16x8*)(Vp + (long long)vr * A + vc8);
  bf16x8 rb = *(const bf16x8*)(Vp + (long long)(vr + 1) * A + vc8);

  wait_async();
  __syncthreads();
  bf16x16 qa0 = load_a(qs + wave * 16 * 64, 64);
  bf16x16 qa1 = load_a(qs + wave * 16 * 64 + 32, 64);

  f32x8 accO[4];
#pragma unroll
  for (int t = 0; t < 4; ++t) accO[t] = {};
  float mrow[8], lrow[8];
#pragma unroll
  for (int r = 0; r < 8; ++r) { mrow[r] = -3.0e38f; lrow[r] = 0.f; }
  int qrow = q0 + wave * 16 + g * 8;

  constexpr int T = S / 64;
  for (int i = 0; i < T; ++i) {
    int kb = i * 64;
    __bf16* ktc = kt[i & 1];
    __bf16* vtc = vt[i & 1];
    wait_async();                        // tile i's K async copies done
    // store prefetched V regs transposed -> vt (packed b32)
#pragma unroll
    for (int jj = 0; jj < 8; ++jj) {
      union { __bf16 h[2]; unsigned u; } pk;
      pk.h[0] = ra[jj]; pk.h[1] = rb[jj];
      *(unsigned*)(vtc + (vc8 + jj) * 64 + vr) = pk.u;
    }
    __syncthreads();                     // tile i fully in LDS; prior buffer free
    if (i + 1 < T) {                     // overlap: stage tile i+1
      __bf16* ktn = kt[(i + 1) & 1];
      const __bf16* Ktile = Kp + (long long)(kb + 64) * A;
#pragma unroll
      for (int j = 0; j < 2; ++j) {
        int c = tid + 256 * j;
        cp_async16(ktn + c * 8, Ktile + c * 8);
      }
      ra = *(const bf16x8*)(Vp + (long long)(kb + 64 + vr) * A + vc8);
      rb = *(const bf16x8*)(Vp + (long long)(kb + 64 + vr + 1) * A + vc8);
    }

    f32x8 s[4];                          // scores 16x64
#pragma unroll
    for (int t = 0; t < 4; ++t) {
      f32x8 c = {};
      c = wmma_bf16(qa0, load_bt(ktc + t * 16 * 64, 64), c);
      c = wmma_bf16(qa1, load_bt(ktc + t * 16 * 64 + 32, 64), c);
      s[t] = c;
    }

    float rmax[8];
#pragma unroll
    for (int r = 0; r < 8; ++r) rmax[r] = -3.0e38f;
#pragma unroll
    for (int t = 0; t < 4; ++t) {
      int kcol = kb + t * 16 + nl;
#pragma unroll
      for (int r = 0; r < 8; ++r) {
        float v = s[t][r];
        v = (kcol <= qrow + r) ? -1.0e9f : v;   // faithful: mask k <= q
        s[t][r] = v;
        rmax[r] = fmaxf(rmax[r], v);
      }
    }
#pragma unroll
    for (int r = 0; r < 8; ++r) {
#pragma unroll
      for (int off = 1; off < 16; off <<= 1)    // row lives in a 16-lane half
        rmax[r] = fmaxf(rmax[r], __shfl_xor(rmax[r], off, 32));
    }
    float scl[8];
#pragma unroll
    for (int r = 0; r < 8; ++r) {
      float mnew = fmaxf(mrow[r], rmax[r]);
      scl[r] = __expf(mrow[r] - mnew);
      mrow[r] = mnew;
    }
    float rs[8];
#pragma unroll
    for (int r = 0; r < 8; ++r) rs[r] = 0.f;
#pragma unroll
    for (int t = 0; t < 4; ++t) {
#pragma unroll
      for (int r = 0; r < 8; ++r) {
        float p = __expf(s[t][r] - mrow[r]);
        rs[r] += p;
        ps[wave][(g * 8 + r) * 64 + t * 16 + nl] = (__bf16)p;
      }
    }
#pragma unroll
    for (int r = 0; r < 8; ++r) {
#pragma unroll
      for (int off = 1; off < 16; off <<= 1)
        rs[r] += __shfl_xor(rs[r], off, 32);
      lrow[r] = lrow[r] * scl[r] + rs[r];
#pragma unroll
      for (int t = 0; t < 4; ++t) accO[t][r] *= scl[r];
    }
    // accO += P @ V (ps wave-private; DS ops in-order per wave)
#pragma unroll
    for (int ks = 0; ks < 2; ++ks) {
      bf16x16 pa = load_a(ps[wave] + ks * 32, 64);
#pragma unroll
      for (int t = 0; t < 4; ++t)
        accO[t] = wmma_bf16(pa, load_bt(vtc + t * 16 * 64 + ks * 32, 64), accO[t]);
    }
  }

  __bf16* Op = O + base;
#pragma unroll
  for (int t = 0; t < 4; ++t) {
#pragma unroll
    for (int r = 0; r < 8; ++r) {
      float v = accO[t][r] / lrow[r];
      Op[(long long)(qrow + r) * A + t * 16 + nl] = (__bf16)v;
    }
  }
}

// ---------------- Kernel 3: out = concat(O) @ Wo + bo  (fp32 out)
DEV void k3_load(const __bf16* O, const float* Wo, int rowbase, int nbase, int kb,
                 int tid, bf16x8* orow, float (*wr)[4]) {
#pragma unroll
  for (int j = 0; j < 2; ++j) {
    int c = tid + 256 * j;               // 512 chunks of 8 bf16
    int r = c >> 2, k = (c & 3) * 8;
    int row = rowbase + r;
    int b = row >> 11, s = row & 2047;
    int f = kb + k;                      // stays inside one head
    int h = f >> 6, a = f & 63;
    orow[j] = *(const bf16x8*)(O + (((long long)b * H + h) * S + s) * A + a);
  }
#pragma unroll
  for (int j = 0; j < 4; ++j) {
    int gi = tid + 256 * j;              // 1024 groups, 4 strided along k
    int n = gi >> 3, k = (gi & 7) * 4;
    const float* wp = Wo + (long long)(kb + k) * D + nbase + n;
    wr[j][0] = wp[0]; wr[j][1] = wp[D]; wr[j][2] = wp[2 * D]; wr[j][3] = wp[3 * D];
  }
}

__global__ __launch_bounds__(256)
void out_proj_kernel(const __bf16* __restrict__ O, const float* __restrict__ Wo,
                     const float* __restrict__ bo, float* __restrict__ out) {
  __shared__ __align__(16) __bf16 xs[2][128 * 32];
  __shared__ __align__(16) __bf16 wt[2][128 * 32];
  int tid = threadIdx.x, lane = tid & 31, wave = tid >> 5;
  int nl = lane & 15, g = lane >> 4;
  int rowbase = blockIdx.x * 128;        // over B*S
  int nbase = blockIdx.y * 128;          // over D

  f32x8 acc[8];
#pragma unroll
  for (int t = 0; t < 8; ++t) acc[t] = {};

  bf16x8 orow[2];
  float wr[4][4];
  k3_load(O, Wo, rowbase, nbase, 0, tid, orow, wr);

  constexpr int NC = (H * A) / 32;
  for (int c = 0; c < NC; ++c) {
    __bf16* xsc = xs[c & 1];
    __bf16* wtc = wt[c & 1];
#pragma unroll
    for (int j = 0; j < 2; ++j) {
      int cc = tid + 256 * j;
      int r = cc >> 2, k = (cc & 3) * 8;
      *(bf16x8*)(xsc + r * 32 + k) = orow[j];
    }
#pragma unroll
    for (int j = 0; j < 4; ++j) {
      int gi = tid + 256 * j;
      int n = gi >> 3, k = (gi & 7) * 4;
      unsigned p0 = pack_bf16(wr[j][0], wr[j][1]);
      unsigned p1 = pack_bf16(wr[j][2], wr[j][3]);
      *(unsigned long long*)(wtc + n * 32 + k) =
          ((unsigned long long)p1 << 32) | p0;
    }
    __syncthreads();
    if (c + 1 < NC) k3_load(O, Wo, rowbase, nbase, (c + 1) * 32, tid, orow, wr);
    bf16x16 af = load_a(xsc + wave * 16 * 32, 32);
#pragma unroll
    for (int t = 0; t < 8; ++t)
      acc[t] = wmma_bf16(af, load_bt(wtc + t * 16 * 32, 32), acc[t]);
  }

#pragma unroll
  for (int t = 0; t < 8; ++t) {
    int col = nbase + t * 16 + nl;
    float bias = bo[col];
#pragma unroll
    for (int r = 0; r < 8; ++r) {
      int row = rowbase + wave * 16 + g * 8 + r;
      out[(long long)row * D + col] = acc[t][r] + bias;
    }
  }
}

extern "C" void kernel_launch(void* const* d_in, const int* in_sizes, int n_in,
                              void* d_out, int out_size, void* d_ws, size_t ws_size,
                              hipStream_t stream) {
  const float* query = (const float*)d_in[0];
  const float* key   = (const float*)d_in[1];
  const float* value = (const float*)d_in[2];
  const float* Wq = (const float*)d_in[3];
  const float* bq = (const float*)d_in[4];
  const float* Wk = (const float*)d_in[5];
  const float* bk = (const float*)d_in[6];
  const float* Wv = (const float*)d_in[7];
  const float* bv = (const float*)d_in[8];
  const float* Wo = (const float*)d_in[9];
  const float* bo = (const float*)d_in[10];
  float* out = (float*)d_out;

  // workspace: Q,K,V,O as bf16 [B,H,S,A] — 4 x 8MB = 32MB
  __bf16* Qb = (__bf16*)d_ws;
  __bf16* Kb = Qb + BHSA;
  __bf16* Vb = Kb + BHSA;
  __bf16* Ob = Vb + BHSA;

  dim3 blk(256);
  dim3 g1(S / 128, BH);
  // 1/sqrt(A) = 0.125 folded into Q (exact power of two in bf16)
  qkv_proj_kernel<<<g1, blk, 0, stream>>>(query, Wq, bq, Qb, 0.125f);
  qkv_proj_kernel<<<g1, blk, 0, stream>>>(key,   Wk, bk, Kb, 1.0f);
  qkv_proj_kernel<<<g1, blk, 0, stream>>>(value, Wv, bv, Vb, 1.0f);
  attn_kernel<<<dim3(S / 128, BH), blk, 0, stream>>>(Qb, Kb, Vb, Ob);
  out_proj_kernel<<<dim3((BB * S) / 128, D / 128), blk, 0, stream>>>(Ob, Wo, bo, out);
}
